// LinearAttention_76270029242753
// MI455X (gfx1250) — compile-verified, matrix operand order checked by reading, not run on hardware
//
#include <hip/hip_runtime.h>
#include <hip/hip_bf16.h>

// ---------------------------------------------------------------------------
// Linear attention on MI455X (gfx1250), wave32 + WMMA bf16 pipeline.
//   B=8, N=4096, DIM=512, HEADS=8, DIM_HEAD=64, HIDDEN=512
// QKV GEMM: 64x128 block tile, double-buffered TDM (tensor_load_to_lds)
//           staging, 8 waves x 4 accumulators, v_wmma_f32_16x16x32_bf16,
//           vectorized bf16x8 epilogue stores.
// ---------------------------------------------------------------------------

#define BATCH    8
#define SEQ      4096
#define DIMM     512
#define HEADS    8
#define DHEAD    64
#define HIDDEN   512          // HEADS * DHEAD
#define QKV3     1536         // 3 * HIDDEN
#define SCALE_F  0.125f       // DHEAD^-0.5
#define NEG_F    (-1.0e15f)

typedef __attribute__((ext_vector_type(16))) __bf16 bf16x16;
typedef __attribute__((ext_vector_type(8)))  __bf16 bf16x8;
typedef __attribute__((ext_vector_type(4)))  __bf16 bf16x4;
typedef __attribute__((ext_vector_type(8)))  float  f32x8;
typedef __attribute__((ext_vector_type(4)))  unsigned int u32x4;
typedef __attribute__((ext_vector_type(8)))  int    i32x8;
typedef __attribute__((ext_vector_type(4)))  int    i32x4;

#if __has_builtin(__builtin_amdgcn_tensor_load_to_lds)
#define HAVE_TDM 1
#else
#define HAVE_TDM 0
#endif

__device__ __forceinline__ f32x8 wmma_bf16(bf16x16 a, bf16x16 b, f32x8 c) {
    // (neg_a, A, neg_b, B, c_mod, C, reuse_a, reuse_b)
    return __builtin_amdgcn_wmma_f32_16x16x32_bf16(false, a, false, b,
                                                   (short)0, c, false, false);
}

// K-pair base within a 16x32 (or 32x16) bf16 fragment, per ISA 7.12.2:
// VGPR v, lane half (0: lanes 0-15, 1: lanes 16-31)
__device__ __forceinline__ int frag_kbase(int v, int half) {
    return ((v >> 2) << 4) + (half << 3) + ((v & 3) << 1);
}

// Load one 16x32 bf16 fragment from a row-major LDS tile (32 elems per row).
// Lane owns row (lane&15)+row0; its 16 elements live at byte chunks
// [half*16, +16) and [32+half*16, +16) of that row -> two ds_load_b128.
__device__ __forceinline__ bf16x16 frag_from_lds(const __bf16* tile,
                                                 int row0, int lane) {
    const int half = lane >> 4;
    const __bf16* r = tile + (size_t)(row0 + (lane & 15)) * 32;
    bf16x8 lo = *(const bf16x8*)(r + half * 8);
    bf16x8 hi = *(const bf16x8*)(r + 16 + half * 8);
    return __builtin_shufflevector(lo, hi, 0, 1, 2, 3, 4, 5, 6, 7,
                                   8, 9, 10, 11, 12, 13, 14, 15);
}

#if HAVE_TDM
// Generic-pointer low 32 bits == LDS byte offset (flat aperture mapping).
__device__ __forceinline__ unsigned lds_off(const void* p) {
    return (unsigned)(unsigned long long)p;
}

// Issue one TDM 2-D tile load: tile_h rows x tile_w bf16 elems,
// row stride `stride` elems, packed row-major into LDS at `lds`.
// D# layout per CDNA5 ISA ch.8 (data_size=1 -> 2 bytes, type=2).
__device__ __forceinline__ void tdm_load_tile_bf16(unsigned lds,
                                                   const __bf16* gptr,
                                                   unsigned tile_w, unsigned tile_h,
                                                   unsigned tensor_w, unsigned tensor_h,
                                                   unsigned stride) {
    unsigned long long ga = (unsigned long long)gptr;
    u32x4 g0;
    g0[0] = 1u;                                        // count=1 user D#
    g0[1] = lds;                                       // lds_addr
    g0[2] = (unsigned)(ga & 0xFFFFFFFFu);              // global_addr[31:0]
    g0[3] = (unsigned)((ga >> 32) & 0x01FFFFFFu)       // global_addr[56:32]
            | 0x80000000u;                             // type=2 at bits[127:126]
    i32x8 g1;
    g1[0] = 0x00010000;                                // data_size=1 (2B), mask=0
    g1[1] = (int)(tensor_w << 16);                     // tensor_dim0[15:0]
    g1[2] = (int)((tensor_w >> 16) | (tensor_h << 16));// dim0 hi | dim1 lo
    g1[3] = (int)((tensor_h >> 16) | (tile_w << 16));  // dim1 hi | tile_dim0
    g1[4] = (int)(tile_h & 0xFFFFu);                   // tile_dim1 (tile_dim2=0)
    g1[5] = (int)stride;                               // tensor_dim0_stride lo32
    g1[6] = 0;                                         // stride hi | dim1_stride lo
    g1[7] = 0;
    i32x4 z4 = {0, 0, 0, 0};
#if defined(__clang_major__) && __clang_major__ >= 23
    i32x8 z8 = {0, 0, 0, 0, 0, 0, 0, 0};
    __builtin_amdgcn_tensor_load_to_lds(g0, g1, z4, z4, z8, 0);
#else
    __builtin_amdgcn_tensor_load_to_lds(g0, g1, z4, z4, 0);
#endif
}
#endif // HAVE_TDM

// ---------------------------------------------------------------------------
// Kernel 0a: convert weights f32 -> bf16 (w_qkv: 1536x512, w_out: 512x512)
// ---------------------------------------------------------------------------
__global__ void la_cvt_weights(const float* __restrict__ wq,
                               const float* __restrict__ wo,
                               __bf16* __restrict__ wqb,
                               __bf16* __restrict__ wob) {
    size_t i = (size_t)blockIdx.x * blockDim.x + threadIdx.x;
    const size_t nq = (size_t)QKV3 * DIMM;
    const size_t no = (size_t)DIMM * HIDDEN;
    if (i < nq) wqb[i] = (__bf16)wq[i];
    if (i < no) wob[i] = (__bf16)wo[i];
}

// ---------------------------------------------------------------------------
// Kernel 0b: convert activations f32 -> bf16 once (halves QKV-GEMM A traffic).
// ---------------------------------------------------------------------------
__global__ void la_cvt_x(const float* __restrict__ x, __bf16* __restrict__ xb) {
    const size_t i = ((size_t)blockIdx.x * blockDim.x + threadIdx.x) * 4;
    const float4 f = *(const float4*)(x + i);
    bf16x4 o = {(__bf16)f.x, (__bf16)f.y, (__bf16)f.z, (__bf16)f.w};
    *(bf16x4*)(xb + i) = o;
}

// ---------------------------------------------------------------------------
// Kernel 1: QKV GEMM.  qkv[row,e] = sum_d xb[row,d] * w_qkv[e,d]
// Block (256 thr = 8 waves) computes a 64(row) x 128(e) tile.
// Double-buffered TDM: while waves compute from buf[cur], wave 0 DMAs the
// next 64x32 / 128x32 bf16 k-slab into buf[cur^1]; s_wait_tensorcnt(0) +
// barrier at the end of each step. Wave w: m_sub = w&3, e_subs = (w>>2)*4..+3.
// Epilogue: per t, 8 rows land at consecutive n -> one bf16x8 store each,
// with uniform Q/K/V routing and length masking (Q=0, K=NEG, V=0 at mask).
// Grid: (32768/64) * (1536/128) = 512 * 12 = 6144 blocks.
// ---------------------------------------------------------------------------
__global__ void la_qkv_gemm(const __bf16* __restrict__ xb,
                            const __bf16* __restrict__ wqb,
                            const int* __restrict__ lengths,
                            __bf16* __restrict__ Qb,
                            __bf16* __restrict__ Kb,
                            __bf16* __restrict__ Vb) {
    __shared__ __bf16 lds_a[2][64 * 32];    // 2 x 4 KB
    __shared__ __bf16 lds_b[2][128 * 32];   // 2 x 8 KB

    const int lane  = threadIdx.x & 31;
    const int wave  = threadIdx.x >> 5;
    const int half  = lane >> 4;
    const int m_blk = blockIdx.x / 12;   // 0..511  (64 rows each)
    const int e_blk = blockIdx.x % 12;   // 0..11   (128 cols each)

    const int m_sub = wave & 3;          // 16-row subtile
    const int esub0 = (wave >> 2) * 4;   // first of 4 16-col subtiles

    const __bf16* Arow0 = xb  + (size_t)m_blk * 64 * DIMM;
    const __bf16* Brow0 = wqb + (size_t)e_blk * 128 * DIMM;

    f32x8 acc[4] = {};

#if HAVE_TDM
    // ---- prologue: fill buffer 0 ----
    if (threadIdx.x < 32) {
        tdm_load_tile_bf16(lds_off(lds_a[0]), Arow0,
                           /*tile*/32, 64, /*tensor*/DIMM, 64, DIMM);
        tdm_load_tile_bf16(lds_off(lds_b[0]), Brow0,
                           /*tile*/32, 128, /*tensor*/DIMM, 128, DIMM);
        __builtin_amdgcn_s_wait_tensorcnt(0);
    }
    __syncthreads();

    int cur = 0;
    for (int kk = 0; kk < DIMM; kk += 32) {
        // ---- prefetch next k-slab into the alternate buffer ----
        if (kk + 32 < DIMM && threadIdx.x < 32) {
            tdm_load_tile_bf16(lds_off(lds_a[cur ^ 1]), Arow0 + kk + 32,
                               32, 64, DIMM, 64, DIMM);
            tdm_load_tile_bf16(lds_off(lds_b[cur ^ 1]), Brow0 + kk + 32,
                               32, 128, DIMM, 128, DIMM);
        }
        // ---- compute on current buffer (overlaps the DMA) ----
        const bf16x16 afrag = frag_from_lds(lds_a[cur], m_sub * 16, lane);
#pragma unroll
        for (int t = 0; t < 4; ++t) {
            const bf16x16 bfrag = frag_from_lds(lds_b[cur], (esub0 + t) * 16, lane);
            acc[t] = wmma_bf16(afrag, bfrag, acc[t]);
        }
        // ---- next buffer ready + all waves done reading cur ----
        if (threadIdx.x < 32) __builtin_amdgcn_s_wait_tensorcnt(0);
        __syncthreads();
        cur ^= 1;
    }
#else
    for (int kk = 0; kk < DIMM; kk += 32) {
        __syncthreads();
        {   // cooperative copy: A = 256 x 16B chunks, B = 512 x 16B chunks
            int c = threadIdx.x;
            int row = c >> 2, cw = c & 3;
            *(uint4*)(lds_a[0] + row * 32 + cw * 8) =
                *(const uint4*)(Arow0 + (size_t)row * DIMM + kk + cw * 8);
#pragma unroll
            for (int j = 0; j < 2; ++j) {
                int cb = threadIdx.x + j * 256;
                int rb = cb >> 2, cwb = cb & 3;
                *(uint4*)(lds_b[0] + rb * 32 + cwb * 8) =
                    *(const uint4*)(Brow0 + (size_t)rb * DIMM + kk + cwb * 8);
            }
        }
        __syncthreads();
        const bf16x16 afrag = frag_from_lds(lds_a[0], m_sub * 16, lane);
#pragma unroll
        for (int t = 0; t < 4; ++t) {
            const bf16x16 bfrag = frag_from_lds(lds_b[0], (esub0 + t) * 16, lane);
            acc[t] = wmma_bf16(afrag, bfrag, acc[t]);
        }
    }
#endif

    // ---- epilogue: vectorized scatter with masking ----
    const int b   = (m_blk * 64) >> 12;          // 64 | 4096: no batch straddle
    const int len = lengths[b];
    const int n0  = (m_blk * 64 + m_sub * 16 + (half << 3)) & (SEQ - 1);
#pragma unroll
    for (int t = 0; t < 4; ++t) {
        const int e = e_blk * 128 + (esub0 + t) * 16 + (lane & 15);
        bf16x8 pk;
        if (e < HIDDEN) {                         // Q (scaled, mask -> 0)
#pragma unroll
            for (int r = 0; r < 8; ++r)
                pk[r] = (__bf16)((n0 + r >= len) ? 0.f : acc[t][r] * SCALE_F);
            const int h = e >> 6, d = e & 63;
            *(bf16x8*)(Qb + (((size_t)b * HEADS + h) * DHEAD + d) * SEQ + n0) = pk;
        } else if (e < 2 * HIDDEN) {              // K (mask -> NEG pre-softmax)
#pragma unroll
            for (int r = 0; r < 8; ++r)
                pk[r] = (__bf16)((n0 + r >= len) ? NEG_F : acc[t][r]);
            const int ek = e - HIDDEN;
            const int h = ek >> 6, d = ek & 63;
            *(bf16x8*)(Kb + (((size_t)b * HEADS + h) * DHEAD + d) * SEQ + n0) = pk;
        } else {                                  // V (mask -> 0)
#pragma unroll
            for (int r = 0; r < 8; ++r)
                pk[r] = (__bf16)((n0 + r >= len) ? 0.f : acc[t][r]);
            const int ev = e - 2 * HIDDEN;
            const int h = ev >> 6, d = ev & 63;
            *(bf16x8*)(Vb + (((size_t)b * HEADS + h) * DHEAD + d) * SEQ + n0) = pk;
        }
    }
}

// ---------------------------------------------------------------------------
// Kernel 2: softmax over N per (b,h,d) row of Kb, in place (bf16 io, f32 math).
// ---------------------------------------------------------------------------
__global__ void la_softmax_rows(__bf16* __restrict__ Kb) {
    const int row = blockIdx.x;
    __bf16* p = Kb + (size_t)row * SEQ;
    const int t = threadIdx.x;

    float vals[16];
    float mx = -3.4e38f;
#pragma unroll
    for (int i = 0; i < 16; ++i) {
        const float f = (float)p[t + i * 256];
        vals[i] = f;
        mx = fmaxf(mx, f);
    }
    __shared__ float red[256];
    red[t] = mx;
    __syncthreads();
    for (int s = 128; s > 0; s >>= 1) {
        if (t < s) red[t] = fmaxf(red[t], red[t + s]);
        __syncthreads();
    }
    const float rmax = red[0];
    __syncthreads();

    float sum = 0.f;
#pragma unroll
    for (int i = 0; i < 16; ++i) {
        const float e = __expf(vals[i] - rmax);
        vals[i] = e;
        sum += e;
    }
    red[t] = sum;
    __syncthreads();
    for (int s = 128; s > 0; s >>= 1) {
        if (t < s) red[t] += red[t + s];
        __syncthreads();
    }
    const float inv = 1.f / red[0];
#pragma unroll
    for (int i = 0; i < 16; ++i) p[t + i * 256] = (__bf16)(vals[i] * inv);
}

// ---------------------------------------------------------------------------
// Kernel 3: context[b,h,d,e] = sum_n Ksm[d,n] * V[e,n]   (64x64, K=4096)
// Stored TRANSPOSED as Cb[bh][e][d] so stage 4 reads contiguous K.
// ---------------------------------------------------------------------------
__global__ void la_context_gemm(const __bf16* __restrict__ Kb,
                                const __bf16* __restrict__ Vb,
                                __bf16* __restrict__ Cb) {
    const int lane = threadIdx.x & 31;
    const int wave = threadIdx.x >> 5;
    const int tile = blockIdx.x * 4 + wave;

    const int e_tile = tile & 3;
    const int d_tile = (tile >> 2) & 3;
    const int bh     = tile >> 4;

    const __bf16* Kp = Kb + (size_t)bh * DHEAD * SEQ;
    const __bf16* Vp = Vb + (size_t)bh * DHEAD * SEQ;

    const int half = lane >> 4;
    const __bf16* arow = Kp + (size_t)(d_tile * 16 + (lane & 15)) * SEQ;
    const __bf16* brow = Vp + (size_t)(e_tile * 16 + (lane & 15)) * SEQ;

    f32x8 acc = {};
    for (int kk = 0; kk < SEQ; kk += 32) {
        if (kk + 64 < SEQ) __builtin_prefetch(arow + kk + 64, 0, 3);
        bf16x16 a, bm;
#pragma unroll
        for (int v = 0; v < 8; ++v) {
            const int kb = frag_kbase(v, half);
            a[2 * v]      = arow[kk + kb];
            a[2 * v + 1]  = arow[kk + kb + 1];
            bm[2 * v]     = brow[kk + kb];
            bm[2 * v + 1] = brow[kk + kb + 1];
        }
        acc = wmma_bf16(a, bm, acc);
    }
    // 8 outputs land at consecutive d -> one bf16x8 store
    const int e  = e_tile * 16 + (lane & 15);
    const int d0 = d_tile * 16 + (half << 3);
    bf16x8 pk;
#pragma unroll
    for (int r = 0; r < 8; ++r) pk[r] = (__bf16)acc[r];
    *(bf16x8*)(Cb + (size_t)bh * DHEAD * DHEAD + (size_t)e * DHEAD + d0) = pk;
}

// ---------------------------------------------------------------------------
// Kernel 4: out2[b,h,e,n] = sum_d context[d,e] * q[d,n]   (64 x 4096, K=64)
// Stored as O2b[(b,n,c)] with c = h*64+e so stage 5 reads contiguous K.
// ---------------------------------------------------------------------------
__global__ void la_attnout_gemm(const __bf16* __restrict__ Cb,
                                const __bf16* __restrict__ Qb,
                                __bf16* __restrict__ O2b) {
    const int lane = threadIdx.x & 31;
    const int wave = threadIdx.x >> 5;
    const int tile = blockIdx.x * 4 + wave;

    const int n_tile = tile & 255;
    const int e_tile = (tile >> 8) & 3;
    const int bh     = tile >> 10;
    const int b = bh >> 3, h = bh & 7;

    const __bf16* Cp = Cb + (size_t)bh * DHEAD * DHEAD;  // (e,d)
    const __bf16* Qp = Qb + (size_t)bh * DHEAD * SEQ;    // (d,n)

    const int half = lane >> 4;
    const int ncol = n_tile * 16 + (lane & 15);
    const __bf16* arow = Cp + (size_t)(e_tile * 16 + (lane & 15)) * DHEAD;

    f32x8 acc = {};
    for (int kk = 0; kk < DHEAD; kk += 32) {
        bf16x16 a, bm;
#pragma unroll
        for (int v = 0; v < 8; ++v) {
            const int kb = frag_kbase(v, half);
            a[2 * v]      = arow[kk + kb];
            a[2 * v + 1]  = arow[kk + kb + 1];
            bm[2 * v]     = Qp[(size_t)(kk + kb) * SEQ + ncol];
            bm[2 * v + 1] = Qp[(size_t)(kk + kb + 1) * SEQ + ncol];
        }
        acc = wmma_bf16(a, bm, acc);
    }
    // 8 outputs land at consecutive c -> one bf16x8 store
    const int n  = n_tile * 16 + (lane & 15);
    const int c0 = h * DHEAD + e_tile * 16 + (half << 3);
    bf16x8 pk;
#pragma unroll
    for (int r = 0; r < 8; ++r) pk[r] = (__bf16)acc[r];
    *(bf16x8*)(O2b + ((size_t)b * SEQ + n) * HIDDEN + c0) = pk;
}

// ---------------------------------------------------------------------------
// Kernel 5: y[b,n,dout] = sum_c out2[b,n,c] * w_out[dout,c] + b_out[dout]
// ---------------------------------------------------------------------------
__global__ void la_out_proj(const __bf16* __restrict__ O2b,
                            const __bf16* __restrict__ Wob,
                            const float* __restrict__ bout,
                            float* __restrict__ y) {
    const int lane = threadIdx.x & 31;
    const int wave = threadIdx.x >> 5;
    const int tile = blockIdx.x * 4 + wave;

    const int d_tile = tile & 31;
    const int m_tile = tile >> 5;

    const int half = lane >> 4;
    const __bf16* arow = O2b + (size_t)(m_tile * 16 + (lane & 15)) * HIDDEN;
    const __bf16* brow = Wob + (size_t)(d_tile * 16 + (lane & 15)) * HIDDEN;

    f32x8 acc = {};
    for (int kk = 0; kk < HIDDEN; kk += 32) {
        bf16x16 a, bm;
#pragma unroll
        for (int v = 0; v < 8; ++v) {
            const int kb = frag_kbase(v, half);
            a[2 * v]      = arow[kk + kb];
            a[2 * v + 1]  = arow[kk + kb + 1];
            bm[2 * v]     = brow[kk + kb];
            bm[2 * v + 1] = brow[kk + kb + 1];
        }
        acc = wmma_bf16(a, bm, acc);
    }
#pragma unroll
    for (int r = 0; r < 8; ++r) {
        const size_t row = (size_t)m_tile * 16 + r + (half << 3);
        const int dout = d_tile * 16 + (lane & 15);
        y[row * DIMM + dout] = acc[r] + bout[dout];
    }
}

// ---------------------------------------------------------------------------
// Host-side launcher
// ---------------------------------------------------------------------------
extern "C" void kernel_launch(void* const* d_in, const int* in_sizes, int n_in,
                              void* d_out, int out_size, void* d_ws, size_t ws_size,
                              hipStream_t stream) {
    const float* x     = (const float*)d_in[0];
    const float* w_qkv = (const float*)d_in[1];
    const float* w_out = (const float*)d_in[2];
    const float* b_out = (const float*)d_in[3];
    const int*   lens  = (const int*)d_in[4];
    float* y = (float*)d_out;

    char* ws = (char*)d_ws;
    size_t off = 0;
    __bf16* wqb = (__bf16*)(ws + off); off += (size_t)QKV3 * DIMM * 2;           // 1.5 MB
    __bf16* wob = (__bf16*)(ws + off); off += (size_t)DIMM * HIDDEN * 2;         // 0.5 MB
    __bf16* Xb  = (__bf16*)(ws + off); off += (size_t)BATCH * SEQ * DIMM * 2;    // 32 MB
    __bf16* Qb  = (__bf16*)(ws + off); off += (size_t)BATCH * HIDDEN * SEQ * 2;  // 32 MB
    __bf16* Kb  = (__bf16*)(ws + off); off += (size_t)BATCH * HIDDEN * SEQ * 2;  // 32 MB
    __bf16* Vb  = (__bf16*)(ws + off); off += (size_t)BATCH * HIDDEN * SEQ * 2;  // 32 MB
    __bf16* Cb  = (__bf16*)(ws + off); off += (size_t)BATCH * HEADS * DHEAD * DHEAD * 2;
    __bf16* O2b = (__bf16*)(ws + off); off += (size_t)BATCH * SEQ * HIDDEN * 2;  // 32 MB
    (void)ws_size; (void)in_sizes; (void)n_in; (void)out_size;

    // 0: conversions
    la_cvt_weights<<<(QKV3 * DIMM + 255) / 256, 256, 0, stream>>>(w_qkv, w_out, wqb, wob);
    la_cvt_x<<<(BATCH * SEQ * DIMM / 4 + 255) / 256, 256, 0, stream>>>(x, Xb);

    // 1: QKV GEMM (512 row-blocks x 12 col-blocks, 256 thr)
    la_qkv_gemm<<<512 * 12, 256, 0, stream>>>(Xb, wqb, lens, Qb, Kb, Vb);

    // 2: softmax over sequence per (b,h,d) row
    la_softmax_rows<<<BATCH * HEADS * DHEAD, 256, 0, stream>>>(Kb);

    // 3: context GEMM (1024 tiles, 4 waves/block)
    la_context_gemm<<<1024 / 4, 128, 0, stream>>>(Kb, Vb, Cb);

    // 4: attention-out GEMM (65536 tiles)
    la_attnout_gemm<<<65536 / 4, 128, 0, stream>>>(Cb, Qb, O2b);

    // 5: output projection + bias (65536 tiles)
    la_out_proj<<<65536 / 4, 128, 0, stream>>>(O2b, wob, b_out, y);
}
